// TransformerBlock_10505490006022
// MI455X (gfx1250) — compile-verified, matrix operand order checked by reading, not run on hardware
//
#include <hip/hip_runtime.h>

// Flip to 0 if the gfx1250 assembler rejects the async mnemonics.
#define USE_ASYNC_LDS 1

// ---------------- problem constants (match reference) ----------------
constexpr int cB   = 4;
constexpr int cT   = 2048;
constexpr int cD   = 1024;
constexpr int cH   = 16;
constexpr int cHS  = 64;
constexpr int cDFF = 4 * cD;

typedef __attribute__((ext_vector_type(16))) _Float16 v16h;
typedef __attribute__((ext_vector_type(8)))  _Float16 v8h;
typedef __attribute__((ext_vector_type(4)))  _Float16 v4h;
typedef __attribute__((ext_vector_type(8)))  float    v8f;
typedef __attribute__((ext_vector_type(4)))  float    v4f;

constexpr int PAD = 8;   // f16 row padding: keeps 16B alignment, skews banks
constexpr int RW  = 32 + PAD;

// ---------------- async global->LDS copy (gfx1250, ASYNCcnt) ----------------
// Per-lane 16B DMA: LDS[vdst] = MEM[vaddr].  vdst = low 32 bits of the generic
// __shared__ pointer (LDS aperture keeps the wave-relative offset in [31:0]).
__device__ __forceinline__ void async_copy_b128(void* lds_dst,
                                                const void* gsrc) {
#if USE_ASYNC_LDS
  unsigned lds = (unsigned)(unsigned long long)lds_dst;
  unsigned long long ga = (unsigned long long)gsrc;
  asm volatile("global_load_async_to_lds_b128 %0, %1, off"
               :: "v"(lds), "v"(ga) : "memory");
#else
  *(v8h*)lds_dst = *(const v8h*)gsrc;
#endif
}

__device__ __forceinline__ void async_wait() {
#if USE_ASYNC_LDS
  asm volatile("s_wait_asynccnt 0" ::: "memory");
#endif
}

// ---------------- WMMA fragment load (ISA 7.12.2, wave32) -------------------
// 16-bit A 16x32 layout: lane (half,l16) holds row l16; element e -> k =
// ((e>>3)<<4) + 8*half + (e&7).  For a contiguous 32-f16 k-span at p, that is
// two contiguous 16B runs: p[8*half .. +8) and p[16+8*half .. +8).
__device__ __forceinline__ v16h frag16(const _Float16* p, int half) {
  const v8h lo = *(const v8h*)(p + 8 * half);
  const v8h hi = *(const v8h*)(p + 16 + 8 * half);
  return __builtin_shufflevector(lo, hi, 0, 1, 2, 3, 4, 5, 6, 7,
                                 8, 9, 10, 11, 12, 13, 14, 15);
}

// =====================================================================
// Pre-pass A: elementwise f32 -> f16 (for x)
// =====================================================================
__global__ __launch_bounds__(256, 1)
void cvt_f32_f16_kernel(const float* __restrict__ in,
                        _Float16* __restrict__ out, int n4) {
  int i = blockIdx.x * 256 + threadIdx.x;
  if (i < n4) {
    v4f wf = *(const v4f*)&in[(size_t)i * 4];
    *(v4h*)&out[(size_t)i * 4] = __builtin_convertvector(wf, v4h);
  }
}

// =====================================================================
// Pre-pass B: batched transpose+convert  in[K][N] (f32) -> out[N][K] (f16)
// 32x32 tiles through LDS; batch via blockIdx.z (stride K*N both sides).
// =====================================================================
__global__ __launch_bounds__(256, 1)
void transpose_f32_f16_kernel(const float* __restrict__ in,
                              _Float16* __restrict__ out, int K, int N) {
  const size_t bs = (size_t)K * N;
  const float* ib = in + bs * blockIdx.z;
  _Float16*    ob = out + bs * blockIdx.z;
  __shared__ _Float16 t[32][34];
  const int k0 = blockIdx.x * 32, n0 = blockIdx.y * 32;
  const int c = threadIdx.x & 31, r8 = threadIdx.x >> 5;
#pragma unroll
  for (int i = 0; i < 4; ++i) {
    int r = r8 + 8 * i;
    t[c][r] = (_Float16)ib[(size_t)(k0 + r) * N + n0 + c];   // coalesced read
  }
  __syncthreads();
#pragma unroll
  for (int i = 0; i < 4; ++i) {
    int r = r8 + 8 * i;
    ob[(size_t)(n0 + r) * K + k0 + c] = t[r][c];             // coalesced write
  }
}

// =====================================================================
// Kernel 1: fused QKV projection.  Per (b,h):  X[T,D] @ W*[D,HS] + b*
// Double-buffered async-DMA pipeline: tile k+1 streams into LDS while the
// WMMAs consume tile k; one barrier + one asynccnt wait per k-step.
// V is written in transposed [z][e][t] layout for the attention PV GEMM.
// =====================================================================
__global__ __launch_bounds__(256, 1)
void qkv_kernel(const _Float16* __restrict__ xh,
                const _Float16* __restrict__ WqT, const float* __restrict__ bq,
                const _Float16* __restrict__ WkT, const float* __restrict__ bk,
                const _Float16* __restrict__ WvT, const float* __restrict__ bv,
                _Float16* __restrict__ Qh, _Float16* __restrict__ Kh,
                _Float16* __restrict__ VTg) {
  const int z    = blockIdx.y;          // b*H + h
  const int h    = z & (cH - 1);
  const int mb   = blockIdx.x * 128;
  const int tid  = threadIdx.x;
  const int lane = tid & 31;
  const int wave = tid >> 5;
  const int half = lane >> 4;
  const int l16  = lane & 15;

  const _Float16* Ah  = xh + (size_t)(z >> 4) * cT * cD;
  const _Float16* Wt0 = WqT + (size_t)h * cHS * cD;   // [64][1024] per head
  const _Float16* Wt1 = WkT + (size_t)h * cHS * cD;
  const _Float16* Wt2 = WvT + (size_t)h * cHS * cD;

  __shared__ __align__(16) _Float16 As[2][128][RW];     // 20 KB
  __shared__ __align__(16) _Float16 Bt[2][3][64][RW];   // 30 KB

  v8f acc[3][4] = {};

  auto stage = [&](int kk, int p) {
#pragma unroll
    for (int i = 0; i < 2; ++i) {
      int idx = tid + 256 * i;                 // A: 512 chunks
      int r = idx >> 2, c8 = idx & 3;
      async_copy_b128(&As[p][r][c8 * 8],
                      &Ah[(size_t)(mb + r) * cD + kk + c8 * 8]);
    }
    int r = tid >> 2, c8 = tid & 3;            // B: 256 chunks per matrix
    size_t off = (size_t)r * cD + kk + c8 * 8;
    async_copy_b128(&Bt[p][0][r][c8 * 8], &Wt0[off]);
    async_copy_b128(&Bt[p][1][r][c8 * 8], &Wt1[off]);
    async_copy_b128(&Bt[p][2][r][c8 * 8], &Wt2[off]);
  };

  stage(0, 0);
  for (int k0 = 0; k0 < cD; k0 += 32) {
    const int p = (k0 >> 5) & 1;
    async_wait();          // my tile-k copies landed
    __syncthreads();       // everyone's tile-k visible; tile-(k-1) reads done
    if (k0 + 32 < cD) stage(k0 + 32, p ^ 1);   // DMA next tile during compute
    if (k0 + 64 < cD)
      __builtin_prefetch(&Ah[(size_t)(mb + (tid >> 1)) * cD + k0 + 64]);

    v16h a = frag16(&As[p][wave * 16 + l16][0], half);
    v16h bbuf[3][4];
#pragma unroll
    for (int m = 0; m < 3; ++m)
#pragma unroll
      for (int nt = 0; nt < 4; ++nt)
        bbuf[m][nt] = frag16(&Bt[p][m][nt * 16 + l16][0], half);
#pragma unroll
    for (int m = 0; m < 3; ++m)
#pragma unroll
      for (int nt = 0; nt < 4; ++nt)
        acc[m][nt] = __builtin_amdgcn_wmma_f32_16x16x32_f16(
            false, a, false, bbuf[m][nt], (short)0, acc[m][nt], false, false);
  }

  // Epilogue: bias add, Q scale, f16 store. C/D layout: row = r + 8*half.
#pragma unroll
  for (int nt = 0; nt < 4; ++nt) {
#pragma unroll
    for (int r = 0; r < 8; ++r) {
      int ml = wave * 16 + r + 8 * half;
      int n  = nt * 16 + l16;
      int tg = mb + ml;
      size_t o = ((size_t)z * cT + tg) * cHS + n;
      float qv = acc[0][nt][r] + bq[h * cHS + n];
      float kv = acc[1][nt][r] + bk[h * cHS + n];
      float vv = acc[2][nt][r] + bv[h * cHS + n];
      Qh[o] = (_Float16)(qv * 0.125f);   // 1/sqrt(64)
      Kh[o] = (_Float16)kv;
      VTg[(size_t)z * cHS * cT + (size_t)n * cT + tg] = (_Float16)vv;
    }
  }
}

// =====================================================================
// Kernel 2: causal flash attention.  Block = 64 q rows (4 waves x 16),
// streams 32-key K/V tiles (both async DMA; V pre-transposed in global).
// =====================================================================
__global__ __launch_bounds__(128, 1)
void attn_kernel(const _Float16* __restrict__ Qh,
                 const _Float16* __restrict__ Kh,
                 const _Float16* __restrict__ VTg,
                 _Float16* __restrict__ Oh) {
  const int z    = blockIdx.y;          // b*H + h
  const int h    = z & (cH - 1);
  const int b    = z >> 4;
  const int qb   = blockIdx.x * 64;
  const int tid  = threadIdx.x;
  const int lane = tid & 31;
  const int wave = tid >> 5;
  const int half = lane >> 4;
  const int l16  = lane & 15;
  const int m0   = qb + wave * 16;

  const _Float16* Qz  = Qh + (size_t)z * cT * cHS;
  const _Float16* Kz  = Kh + (size_t)z * cT * cHS;
  const _Float16* VTz = VTg + (size_t)z * cHS * cT;   // [64][T]

  __shared__ __align__(16) _Float16 Ks[32][64 + PAD];      // 4.5 KB
  __shared__ __align__(16) _Float16 Vt[64][RW];            // 5 KB (transposed)
  __shared__ __align__(16) _Float16 Ps[4][16][RW];         // 5 KB

  // Q fragments: 16 rows x 64 e -> two A fragments, 4x global b128 per lane.
  const _Float16* qrow = Qz + (size_t)(m0 + l16) * cHS;
  v16h aq0 = frag16(qrow, half);
  v16h aq1 = frag16(qrow + 32, half);

  float mst[8], lst[8];
#pragma unroll
  for (int r = 0; r < 8; ++r) { mst[r] = -3.0e38f; lst[r] = 0.0f; }
  v8f accO[4] = {};

  const int nkb = (qb + 64) / 32;       // causal bound: keys <= last q row
  for (int kb = 0; kb < nkb; ++kb) {
    // --- stage K (32x64 rows) and V (64 e-rows x 32 keys), both async -----
#pragma unroll
    for (int i = 0; i < 2; ++i) {
      int idx = tid + 128 * i;                 // K: 256 chunks
      int r = idx >> 3, c8 = idx & 7;
      async_copy_b128(&Ks[r][c8 * 8],
                      &Kz[((size_t)kb * 32 + r) * cHS + c8 * 8]);
    }
#pragma unroll
    for (int i = 0; i < 2; ++i) {
      int idx = tid + 128 * i;                 // V: 256 chunks
      int r = idx >> 2, c8 = idx & 3;
      async_copy_b128(&Vt[r][c8 * 8],
                      &VTz[(size_t)r * cT + kb * 32 + c8 * 8]);
    }
    async_wait();
    __syncthreads();

    // S = Q * K^T : B fragment = A-style load of K rows (keys).
    v8f s0 = {}, s1 = {};
    {
      v16h kf00 = frag16(&Ks[l16][0], half);
      v16h kf01 = frag16(&Ks[l16][32], half);
      v16h kf10 = frag16(&Ks[16 + l16][0], half);
      v16h kf11 = frag16(&Ks[16 + l16][32], half);
      s0 = __builtin_amdgcn_wmma_f32_16x16x32_f16(false, aq0, false, kf00, (short)0, s0, false, false);
      s0 = __builtin_amdgcn_wmma_f32_16x16x32_f16(false, aq1, false, kf01, (short)0, s0, false, false);
      s1 = __builtin_amdgcn_wmma_f32_16x16x32_f16(false, aq0, false, kf10, (short)0, s1, false, false);
      s1 = __builtin_amdgcn_wmma_f32_16x16x32_f16(false, aq1, false, kf11, (short)0, s1, false, false);
    }

    // Online softmax. Row r+8*half is spread over 16 lanes of this half.
#pragma unroll
    for (int r = 0; r < 8; ++r) {
      int mg  = m0 + r + 8 * half;
      int sg0 = kb * 32 + l16;
      int sg1 = kb * 32 + 16 + l16;
      float v0 = (sg0 <= mg) ? s0[r] : -3.0e38f;
      float v1 = (sg1 <= mg) ? s1[r] : -3.0e38f;
      float rm = fmaxf(v0, v1);
      rm = fmaxf(rm, __shfl_xor(rm, 1));
      rm = fmaxf(rm, __shfl_xor(rm, 2));
      rm = fmaxf(rm, __shfl_xor(rm, 4));
      rm = fmaxf(rm, __shfl_xor(rm, 8));
      float mnew  = fmaxf(mst[r], rm);
      float alpha = __expf(mst[r] - mnew);
      float p0 = __expf(v0 - mnew);
      float p1 = __expf(v1 - mnew);
      float rs = p0 + p1;
      rs += __shfl_xor(rs, 1);
      rs += __shfl_xor(rs, 2);
      rs += __shfl_xor(rs, 4);
      rs += __shfl_xor(rs, 8);
      lst[r] = lst[r] * alpha + rs;
      mst[r] = mnew;
#pragma unroll
      for (int nt = 0; nt < 4; ++nt) accO[nt][r] *= alpha;
      Ps[wave][r + 8 * half][l16]      = (_Float16)p0;
      Ps[wave][r + 8 * half][16 + l16] = (_Float16)p1;
    }
    __syncthreads();   // make P visible across lanes (all waves iterate alike)

    // O += P * V
    v16h ap = frag16(&Ps[wave][l16][0], half);
    v16h bvv[4];
#pragma unroll
    for (int nt = 0; nt < 4; ++nt) bvv[nt] = frag16(&Vt[nt * 16 + l16][0], half);
#pragma unroll
    for (int nt = 0; nt < 4; ++nt)
      accO[nt] = __builtin_amdgcn_wmma_f32_16x16x32_f16(
          false, ap, false, bvv[nt], (short)0, accO[nt], false, false);
    __syncthreads();   // done reading Vt/Ps before next staging
  }

  // Normalize and store O as f16 in [B, T, H*HS] layout for the O-projection.
#pragma unroll
  for (int nt = 0; nt < 4; ++nt) {
#pragma unroll
    for (int r = 0; r < 8; ++r) {
      int mg = m0 + r + 8 * half;
      int n  = nt * 16 + l16;
      float val = accO[nt][r] / lst[r];
      Oh[((size_t)b * cT + mg) * cD + h * cHS + n] = (_Float16)val;
    }
  }
}

// =====================================================================
// Kernels 3-5: generic WMMA GEMM  C = A(f16)[M,K] * BT(f16)[N][K]
// macro tile 128x64, BK=32, 8 waves; double-buffered async-DMA pipeline.
//   EPI 0: out_f32 = resid + acc + bias ; out_f16 = (f16)out_f32
//   EPI 1: out_f16 = (f16) gelu_erf(acc + bias)
//   EPI 2: out_f32 = resid + acc + bias
// =====================================================================
template <int EPI>
__global__ __launch_bounds__(256, 1)
void gemm_epi_kernel(const _Float16* __restrict__ A,
                     const _Float16* __restrict__ BT,
                     const float* __restrict__ bias,
                     const float* __restrict__ resid,
                     float* __restrict__ outf,
                     _Float16* __restrict__ outh,
                     int M, int N, int K) {
  const int tid  = threadIdx.x;
  const int lane = tid & 31;
  const int wave = tid >> 5;
  const int half = lane >> 4;
  const int l16  = lane & 15;
  const int nb   = blockIdx.x * 64;
  const int mb   = blockIdx.y * 128;

  __shared__ __align__(16) _Float16 As[2][128][RW];   // 20 KB
  __shared__ __align__(16) _Float16 Bt[2][64][RW];    // 10 KB

  v8f acc[4] = {};

  auto stage = [&](int kk, int p) {
#pragma unroll
    for (int i = 0; i < 2; ++i) {
      int idx = tid + 256 * i;                 // A: 512 chunks
      int r = idx >> 2, c8 = idx & 3;
      async_copy_b128(&As[p][r][c8 * 8],
                      &A[(size_t)(mb + r) * K + kk + c8 * 8]);
    }
    int r = tid >> 2, c8 = tid & 3;            // B: 256 chunks
    async_copy_b128(&Bt[p][r][c8 * 8],
                    &BT[(size_t)(nb + r) * K + kk + c8 * 8]);
  };

  stage(0, 0);
  for (int k0 = 0; k0 < K; k0 += 32) {
    const int p = (k0 >> 5) & 1;
    async_wait();          // my tile-k copies landed
    __syncthreads();       // everyone's tile-k visible; tile-(k-1) reads done
    if (k0 + 32 < K) stage(k0 + 32, p ^ 1);    // DMA next tile during compute
    if (k0 + 64 < K) {
      __builtin_prefetch(&A[(size_t)(mb + (tid >> 1)) * K + k0 + 64]);
      __builtin_prefetch(&BT[(size_t)(nb + (tid >> 2)) * K + k0 + 64]);
    }

    v16h a = frag16(&As[p][wave * 16 + l16][0], half);
    v16h bbuf[4];
#pragma unroll
    for (int nt = 0; nt < 4; ++nt)
      bbuf[nt] = frag16(&Bt[p][nt * 16 + l16][0], half);
#pragma unroll
    for (int nt = 0; nt < 4; ++nt)
      acc[nt] = __builtin_amdgcn_wmma_f32_16x16x32_f16(
          false, a, false, bbuf[nt], (short)0, acc[nt], false, false);
  }

#pragma unroll
  for (int nt = 0; nt < 4; ++nt) {
#pragma unroll
    for (int r = 0; r < 8; ++r) {
      int m = mb + wave * 16 + r + 8 * half;
      int n = nb + nt * 16 + l16;
      size_t o = (size_t)m * N + n;
      float v = acc[nt][r] + bias[n];
      if (EPI == 0) {
        v += resid[o];
        outf[o] = v;
        outh[o] = (_Float16)v;
      } else if (EPI == 1) {
        float g = 0.5f * v * (1.0f + erff(v * 0.70710678118654752f));
        outh[o] = (_Float16)g;
      } else {
        v += resid[o];
        outf[o] = v;
      }
    }
  }
}

// =====================================================================
// Host orchestration
// =====================================================================
extern "C" void kernel_launch(void* const* d_in, const int* in_sizes, int n_in,
                              void* d_out, int out_size, void* d_ws,
                              size_t ws_size, hipStream_t stream) {
  (void)in_sizes; (void)n_in; (void)out_size; (void)ws_size;
  const float* x  = (const float*)d_in[0];
  const float* Wq = (const float*)d_in[1];
  const float* bq = (const float*)d_in[2];
  const float* Wk = (const float*)d_in[3];
  const float* bk = (const float*)d_in[4];
  const float* Wv = (const float*)d_in[5];
  const float* bv = (const float*)d_in[6];
  const float* Wo = (const float*)d_in[7];
  const float* bo = (const float*)d_in[8];
  const float* W1 = (const float*)d_in[9];
  const float* b1 = (const float*)d_in[10];
  const float* W2 = (const float*)d_in[11];
  const float* b2 = (const float*)d_in[12];

  char* w = (char*)d_ws;
  auto take = [&](size_t bytes) -> char* {
    char* p = w;
    w += (bytes + 255) & ~(size_t)255;
    return p;
  };
  const size_t nBTD = (size_t)cB * cT * cD;          // 8,388,608
  const size_t nHW  = (size_t)cH * cD * cHS;         // 1,048,576 (per QKV mat)
  _Float16* xh   = (_Float16*)take(nBTD * sizeof(_Float16));
  _Float16* WqT  = (_Float16*)take(nHW * sizeof(_Float16));
  _Float16* WkT  = (_Float16*)take(nHW * sizeof(_Float16));
  _Float16* WvT  = (_Float16*)take(nHW * sizeof(_Float16));
  _Float16* WoT  = (_Float16*)take((size_t)cD * cD * sizeof(_Float16));
  _Float16* W1T  = (_Float16*)take((size_t)cD * cDFF * sizeof(_Float16));
  _Float16* W2T  = (_Float16*)take((size_t)cDFF * cD * sizeof(_Float16));
  _Float16* Qh   = (_Float16*)take(nBTD * sizeof(_Float16));
  _Float16* Kh   = (_Float16*)take(nBTD * sizeof(_Float16));
  _Float16* VTg  = (_Float16*)take(nBTD * sizeof(_Float16));
  _Float16* Oh   = (_Float16*)take(nBTD * sizeof(_Float16));
  float*    x1f  = (float*)take(nBTD * sizeof(float));
  _Float16* x1h  = (_Float16*)take(nBTD * sizeof(_Float16));
  _Float16* Hh   = (_Float16*)take((size_t)cB * cT * cDFF * sizeof(_Float16));

  const int M = cB * cT;

  // 0) one-shot conversions/transposes (amortized across all GEMM blocks)
  cvt_f32_f16_kernel<<<dim3((int)(nBTD / 4 / 256)), 256, 0, stream>>>(
      x, xh, (int)(nBTD / 4));
  transpose_f32_f16_kernel<<<dim3(cD / 32, cHS / 32, cH), 256, 0, stream>>>(
      Wq, WqT, cD, cHS);
  transpose_f32_f16_kernel<<<dim3(cD / 32, cHS / 32, cH), 256, 0, stream>>>(
      Wk, WkT, cD, cHS);
  transpose_f32_f16_kernel<<<dim3(cD / 32, cHS / 32, cH), 256, 0, stream>>>(
      Wv, WvT, cD, cHS);
  transpose_f32_f16_kernel<<<dim3(cD / 32, cD / 32, 1), 256, 0, stream>>>(
      Wo, WoT, cD, cD);
  transpose_f32_f16_kernel<<<dim3(cD / 32, cDFF / 32, 1), 256, 0, stream>>>(
      W1, W1T, cD, cDFF);
  transpose_f32_f16_kernel<<<dim3(cDFF / 32, cD / 32, 1), 256, 0, stream>>>(
      W2, W2T, cDFF, cD);

  // 1) QKV projections (Q pre-scaled by 1/sqrt(HS); V stored transposed)
  qkv_kernel<<<dim3(cT / 128, cB * cH), 256, 0, stream>>>(
      xh, WqT, bq, WkT, bk, WvT, bv, Qh, Kh, VTg);

  // 2) causal flash attention
  attn_kernel<<<dim3(cT / 64, cB * cH), 128, 0, stream>>>(Qh, Kh, VTg, Oh);

  // 3) x1 = x + O @ Wo + bo   (store f32 + f16 copy)
  gemm_epi_kernel<0><<<dim3(cD / 64, M / 128), 256, 0, stream>>>(
      Oh, WoT, bo, x, x1f, x1h, M, cD, cD);

  // 4) H = gelu(x1 @ W1 + b1)  (exact erf gelu, stored f16)
  gemm_epi_kernel<1><<<dim3(cDFF / 64, M / 128), 256, 0, stream>>>(
      x1h, W1T, b1, nullptr, nullptr, Hh, M, cDFF, cD);

  // 5) out = x1 + H @ W2 + b2  (f32 to d_out)
  gemm_epi_kernel<2><<<dim3(cD / 64, M / 128), 256, 0, stream>>>(
      Hh, W2T, b2, x1f, (float*)d_out, nullptr, M, cD, cDFF);
}
